// Attention_62302795596087
// MI455X (gfx1250) — compile-verified, matrix operand order checked by reading, not run on hardware
//
#include <hip/hip_runtime.h>

// Flash-attention (causal) for [B=4,H=16,S=2048,d=64] fp32 on gfx1250.
// Tensor path: v_wmma_f32_16x16x32_f16 for both QK^T and PV, fp32 online softmax
// in exp2 domain (log2e folded into the Q scale). Row reductions use DPP16
// (VALU) instead of ds_bpermute.

typedef _Float16 v16h __attribute__((ext_vector_type(16)));
typedef _Float16 v8h  __attribute__((ext_vector_type(8)));
typedef _Float16 v4h  __attribute__((ext_vector_type(4)));
typedef float    v8f  __attribute__((ext_vector_type(8)));

#define S_LEN 2048
#define DHEAD 64
#define QTILE 64       // q rows per block (4 waves x 16)
#define KSTEP 32       // keys per inner step (WMMA K for PV)
#define NWAVE 4
// padded LDS row strides (in halves) to avoid bank conflicts; multiples of 8
#define KPAD  72       // K tile rows: 64 cols -> stride 72 (144B, 16B aligned)
#define VPAD  40       // Vt tile rows: 32 cols -> stride 40 (80B, 16B aligned)
#define PPAD  40       // P tile rows: 32 cols -> stride 40

// ---- DPP16 helpers: data movement confined to each 16-lane row group ----
template <int CTRL>
__device__ inline float dpp_mov(float x) {
  int v = __builtin_amdgcn_update_dpp(0, __float_as_int(x), CTRL, 0xF, 0xF, true);
  return __int_as_float(v);
}
// Full 16-lane butterfly reduce; result replicated to all 16 lanes of the row.
__device__ inline float row16_max(float x) {
  x = fmaxf(x, dpp_mov<0xB1>(x));    // quad_perm(1,0,3,2)  : xor 1
  x = fmaxf(x, dpp_mov<0x4E>(x));    // quad_perm(2,3,0,1)  : xor 2
  x = fmaxf(x, dpp_mov<0x141>(x));   // ROW_HALF_MIRROR     : pair quads
  x = fmaxf(x, dpp_mov<0x140>(x));   // ROW_MIRROR          : pair halves
  return x;
}
__device__ inline float row16_sum(float x) {
  x += dpp_mov<0xB1>(x);
  x += dpp_mov<0x4E>(x);
  x += dpp_mov<0x141>(x);
  x += dpp_mov<0x140>(x);
  return x;
}

// Build a 16x32 (A/B) f16 fragment from two contiguous 8-half (16B) LDS rows.
__device__ inline v16h ld_frag(const _Float16* p_lo, const _Float16* p_hi) {
  v8h lo = *(const v8h*)p_lo;
  v8h hi = *(const v8h*)p_hi;
  v16h r;
#pragma unroll
  for (int i = 0; i < 8; ++i) { r[i] = lo[i]; r[i + 8] = hi[i]; }
  return r;
}

// Load 8 consecutive floats, scale, convert to halves at a[off..off+7].
__device__ inline void load8_scaled(const float* p, float s, v16h& a, int off) {
  float4 u = *(const float4*)p;
  float4 w = *(const float4*)(p + 4);
  a[off + 0] = (_Float16)(u.x * s); a[off + 1] = (_Float16)(u.y * s);
  a[off + 2] = (_Float16)(u.z * s); a[off + 3] = (_Float16)(u.w * s);
  a[off + 4] = (_Float16)(w.x * s); a[off + 5] = (_Float16)(w.y * s);
  a[off + 6] = (_Float16)(w.z * s); a[off + 7] = (_Float16)(w.w * s);
}

__global__ __launch_bounds__(128)
void Attention_62302795596087_kernel(const float* __restrict__ Q,
                                     const float* __restrict__ K,
                                     const float* __restrict__ V,
                                     float* __restrict__ O)
{
  __shared__ alignas(16) _Float16 Klds[KSTEP][KPAD];        // K tile, row-major  (keys x d)
  __shared__ alignas(16) _Float16 Vt[DHEAD][VPAD];          // V tile, transposed (d x keys)
  __shared__ alignas(16) _Float16 Plds[NWAVE][16][PPAD];    // per-wave P tile (rows x keys)

  const int tid  = threadIdx.x;
  const int wave = tid >> 5;
  const int lane = tid & 31;
  const int l16  = lane & 15;
  const int hi   = lane >> 4;          // 0: lanes 0-15, 1: lanes 16-31

  const int qTileIdx = blockIdx.x;     // 0..31
  const int bh       = blockIdx.y;     // 0..63
  const size_t baseBH = (size_t)bh * S_LEN * DHEAD;

  const int qBase0 = qTileIdx * QTILE;
  const int qRow0  = qBase0 + wave * 16;   // first q row owned by this wave
  const int myQRow = qRow0 + l16;          // A-fragment row for this lane

  // ---- Q A-fragments: 16x64 as two 16x32 f16 chunks.
  // Scale folds 1/sqrt(d) AND log2(e): softmax done in exp2 domain (exact).
  const float scale = 0.125f * 1.4426950408889634f;
  v16h aq[2];
  {
    const float* qp = Q + baseBH + (size_t)myQRow * DHEAD;
#pragma unroll
    for (int f = 0; f < 2; ++f) {
      load8_scaled(qp + f * 32 + hi * 8,      scale, aq[f], 0);  // K = 0..7  (lane half)
      load8_scaled(qp + f * 32 + 16 + hi * 8, scale, aq[f], 8);  // K = 16..23 (lane half)
    }
  }

  const float NEG_INF = -__builtin_inff();
  float mrow[8], lrow[8];
  v8f  oacc[4];
#pragma unroll
  for (int i = 0; i < 8; ++i) { mrow[i] = NEG_INF; lrow[i] = 0.0f; }
#pragma unroll
  for (int n = 0; n < 4; ++n) oacc[n] = (v8f)0.0f;

  const int jEnd  = qBase0 + QTILE;  // block-uniform staging bound (causal)
  const int myEnd = qRow0 + 16;      // keys >= myEnd are fully masked for this wave

  for (int j = 0; j < jEnd; j += KSTEP) {
    __syncthreads();   // previous step's LDS consumers are done

    // Prefetch next K/V tiles (locality=3 -> near scope, pulls into all levels)
    if (j + KSTEP < jEnd) {
      __builtin_prefetch(K + baseBH + (size_t)(j + KSTEP) * DHEAD + tid * 16, 0, 3);
      __builtin_prefetch(V + baseBH + (size_t)(j + KSTEP) * DHEAD + tid * 16, 0, 3);
    }

    // ---- cooperative staging: K -> Klds (f16 row-major), V -> Vt (f16 transposed)
    // Issue all 8 global loads first so they overlap, then convert + store.
    float4 kk[4], vv[4];
#pragma unroll
    for (int it = 0; it < 4; ++it) {
      int off = (tid + it * 128) * 4;          // 512 float4 per 32x64 tile
      int row = off >> 6;                      // key within tile
      int col = off & 63;                      // d
      kk[it] = *(const float4*)(K + baseBH + (size_t)(j + row) * DHEAD + col);
      vv[it] = *(const float4*)(V + baseBH + (size_t)(j + row) * DHEAD + col);
    }
#pragma unroll
    for (int it = 0; it < 4; ++it) {
      int off = (tid + it * 128) * 4;
      int row = off >> 6;
      int col = off & 63;
      v4h kh; kh[0] = (_Float16)kk[it].x; kh[1] = (_Float16)kk[it].y;
              kh[2] = (_Float16)kk[it].z; kh[3] = (_Float16)kk[it].w;
      *(v4h*)&Klds[row][col] = kh;
      Vt[col + 0][row] = (_Float16)vv[it].x; Vt[col + 1][row] = (_Float16)vv[it].y;
      Vt[col + 2][row] = (_Float16)vv[it].z; Vt[col + 3][row] = (_Float16)vv[it].w;
    }
    __syncthreads();

    // Wave-uniform skip: everything in this step is masked for this wave.
    if (j >= myEnd) continue;

    // ---- S = Q * K^T : two 16-key groups, 2 WMMAs each (d split 32+32)
    v8f sg[2];
#pragma unroll
    for (int g = 0; g < 2; ++g) {
      v8f c = (v8f)0.0f;
#pragma unroll
      for (int f = 0; f < 2; ++f) {
        v16h kb = ld_frag(&Klds[g * 16 + l16][f * 32 + hi * 8],
                          &Klds[g * 16 + l16][f * 32 + 16 + hi * 8]);
        c = __builtin_amdgcn_wmma_f32_16x16x32_f16(false, aq[f], false, kb,
                                                   (short)0, c, false, false);
      }
      sg[g] = c;
    }

    // ---- causal mask: only the single diagonal step needs it (wave-uniform test)
    if (j + KSTEP - 1 > qRow0) {
#pragma unroll
      for (int i = 0; i < 8; ++i) {
        int qIdx = qRow0 + i + hi * 8;
        int k0   = j + l16;
        if (k0 > qIdx)      sg[0][i] = NEG_INF;
        if (k0 + 16 > qIdx) sg[1][i] = NEG_INF;
      }
    }

    // ---- online softmax in exp2 domain (DPP16 row reductions, no LDS traffic)
    float pr0[8], pr1[8];
#pragma unroll
    for (int i = 0; i < 8; ++i) {
      float t     = row16_max(fmaxf(sg[0][i], sg[1][i]));
      float mnew  = fmaxf(mrow[i], t);
      float alpha = __builtin_exp2f(mrow[i] - mnew);
      float p0 = __builtin_exp2f(sg[0][i] - mnew);
      float p1 = __builtin_exp2f(sg[1][i] - mnew);
      float rs = row16_sum(p0 + p1);
      lrow[i] = lrow[i] * alpha + rs;
      mrow[i] = mnew;
#pragma unroll
      for (int n = 0; n < 4; ++n) oacc[n][i] *= alpha;
      pr0[i] = p0; pr1[i] = p1;
    }

    // ---- relayout P: C-layout -> A-layout via wave-private LDS tile
#pragma unroll
    for (int i = 0; i < 8; ++i) {
      int r = i + hi * 8;
      Plds[wave][r][l16]      = (_Float16)pr0[i];
      Plds[wave][r][l16 + 16] = (_Float16)pr1[i];
    }
    v16h pa = ld_frag(&Plds[wave][l16][hi * 8],
                      &Plds[wave][l16][16 + hi * 8]);

    // ---- O += P * V : 4 d-column tiles of 16
#pragma unroll
    for (int n = 0; n < 4; ++n) {
      v16h bv = ld_frag(&Vt[n * 16 + l16][hi * 8],
                        &Vt[n * 16 + l16][16 + hi * 8]);
      oacc[n] = __builtin_amdgcn_wmma_f32_16x16x32_f16(false, pa, false, bv,
                                                       (short)0, oacc[n], false, false);
    }
  }

  // ---- epilogue: O / l, coalesced fp32 stores (16 lanes cover 16 contiguous cols)
  float* op = O + baseBH;
#pragma unroll
  for (int i = 0; i < 8; ++i) {
    float inv = 1.0f / lrow[i];
    int row = qRow0 + i + hi * 8;
#pragma unroll
    for (int n = 0; n < 4; ++n)
      op[(size_t)row * DHEAD + n * 16 + l16] = oacc[n][i] * inv;
  }
}

extern "C" void kernel_launch(void* const* d_in, const int* in_sizes, int n_in,
                              void* d_out, int out_size, void* d_ws, size_t ws_size,
                              hipStream_t stream) {
  (void)in_sizes; (void)n_in; (void)out_size; (void)d_ws; (void)ws_size;
  const float* q = (const float*)d_in[0];
  const float* k = (const float*)d_in[1];
  const float* v = (const float*)d_in[2];
  float* o = (float*)d_out;
  dim3 grid(S_LEN / QTILE, 4 * 16);   // 32 q-tiles x 64 (batch*head)
  dim3 block(128);                    // 4 waves (wave32)
  Attention_62302795596087_kernel<<<grid, block, 0, stream>>>(q, k, v, o);
}
